// DeltaNetEnhancedAttention_69569880261167
// MI455X (gfx1250) — matrix operand order repaired and checked by reading, not compiled, not run on hardware
//
#include <hip/hip_runtime.h>
#include <hip/hip_bf16.h>

// ---------------------------------------------------------------------------
// DeltaNet enhanced attention for MI455X (gfx1250, wave32, WMMA).
//
// Key optimizations (vs reference):
//  * attn_out is dead code; beta == softmax row-sum == 1  ->  the O(N^2)
//    softmax attention is eliminated entirely.
//  * All large GEMMs (x@w_qkv, x@w_id, x@w_g1, ctx@w_proj) and all delta-rule
//    64x64x64 matmuls run on v_wmma_f32_16x16x32_f16 (f32 accumulate).
//  * GEMM tiles are staged with CDNA5 async global->LDS transfers
//    (global_load_async_to_lds_b128 + s_wait_asynccnt), bypassing VGPRs.
//  * u_t = u - w@S is rewritten as T @ (v - k@S), saving one 64^3 matmul.
// Shapes: B=4, N=1024, C=768, H=12, D=64, chunk=64, P=5.
// ---------------------------------------------------------------------------

typedef __attribute__((ext_vector_type(16))) _Float16 v16h;
typedef __attribute__((ext_vector_type(8)))  float    v8f;

#define WMMA_F16(a, b, c) \
  __builtin_amdgcn_wmma_f32_16x16x32_f16(false, (a), false, (b), (short)0, (c), false, false)

__device__ __forceinline__ int lane_id() { return threadIdx.x & 31; }

// Generic shared-pointer -> 32-bit LDS byte offset (LDS aperture keeps the DS
// offset in the low 32 bits of the generic address).
__device__ __forceinline__ unsigned lds_off32(const void* p) {
  return (unsigned)(unsigned long long)p;
}

// Async 16-byte global -> LDS transfer (one per lane), tracked by ASYNCcnt.
__device__ __forceinline__ void async_copy_b128(void* lds_dst, const void* gsrc) {
  unsigned loff = lds_off32(lds_dst);
  asm volatile("global_load_async_to_lds_b128 %0, %1, off"
               :: "v"(loff), "v"(gsrc) : "memory");
}
__device__ __forceinline__ void async_wait_all() {
  asm volatile("s_wait_asynccnt 0" ::: "memory");
}

// A fragment: 16x32 f16 tile, rows m0..m0+15, cols k0..k0+31 of row-major src.
// Layout: lane l holds row m0+(l&15); element j = K (k0 + half*16 + j)
// (half = l>>4), matching the B fragment order.
__device__ __forceinline__ v16h load_frag_a(const _Float16* src, int ld, int m0, int k0) {
  int l = lane_id(); int m = m0 + (l & 15); int half = l >> 4;
  const _Float16* p = src + (size_t)m * ld + k0 + half * 16;
  v16h a;
#pragma unroll
  for (int j = 0; j < 8; ++j) a[j] = p[j];
#pragma unroll
  for (int j = 0; j < 8; ++j) a[8 + j] = p[8 + j];
  return a;
}

// A fragment of X^T where X is row-major (A[m][k] = X[k][m]).
__device__ __forceinline__ v16h load_frag_at(const _Float16* X, int ld, int m0, int k0) {
  int l = lane_id(); int m = m0 + (l & 15); int half = l >> 4;
  v16h a;
#pragma unroll
  for (int j = 0; j < 8; ++j) a[j]     = X[(size_t)(k0 + half * 16 + j) * ld + m];
#pragma unroll
  for (int j = 0; j < 8; ++j) a[8 + j] = X[(size_t)(k0 + half * 16 + 8 + j) * ld + m];
  return a;
}

// B fragment: 32x16 f16 tile of row-major KxN matrix.
// Layout (contiguous k, same as A): lane l holds col n0+(l&15); element j = K (k0 + half*16 + j).
__device__ __forceinline__ v16h load_frag_b(const _Float16* src, int ld, int k0, int n0) {
  int l = lane_id(); int n = n0 + (l & 15); int half = l >> 4;
  const _Float16* p = src + (size_t)(k0 + half * 16) * ld + n;
  v16h b;
#pragma unroll
  for (int j = 0; j < 16; ++j) b[j] = p[(size_t)j * ld];
  return b;
}

// B fragment of X^T where X is row-major NxK (B[k][n] = X[n][k]) -> contiguous.
__device__ __forceinline__ v16h load_frag_bt(const _Float16* X, int ld, int k0, int n0) {
  int l = lane_id(); int n = n0 + (l & 15); int half = l >> 4;
  const _Float16* p = X + (size_t)n * ld + k0 + half * 16;
  v16h b;
#pragma unroll
  for (int j = 0; j < 16; ++j) b[j] = p[j];
  return b;
}

__device__ __forceinline__ float gelu_tanh(float x) {
  const float c = 0.7978845608028654f;
  float t = tanhf(c * (x + 0.044715f * x * x * x));
  return 0.5f * x * (1.f + t);
}

// ---------------------------------------------------------------------------
// f32 -> f16 conversion
// ---------------------------------------------------------------------------
__global__ void cvt_f16_kernel(const float* __restrict__ s, _Float16* __restrict__ d, int n) {
  int i = blockIdx.x * blockDim.x + threadIdx.x;
  if (i < n) d[i] = (_Float16)s[i];
}

// ---------------------------------------------------------------------------
// Generic f16 WMMA GEMM: C[M,N] = A[M,K] @ B[K,N] (+bias) (+activation)
// Block: 256 threads = 8 waves; block tile 64x64; wave tile 32x16 (2 WMMA
// accumulators sharing one B fragment). M,N multiples of 64; K multiple of 32.
// Tiles staged via async global->LDS (ASYNCcnt).  act: 0 = none, 1 = gelu.
// ---------------------------------------------------------------------------
__global__ __launch_bounds__(256) void gemm_f16_wmma(
    const _Float16* __restrict__ A, const _Float16* __restrict__ B,
    float* __restrict__ C, const float* __restrict__ bias,
    int M, int K, int N, int act) {
  __shared__ _Float16 As[64 * 32];  // [m][k]
  __shared__ _Float16 Bs[32 * 64];  // [k][n]
  int bm = blockIdx.y, bn = blockIdx.x;
  int tid = threadIdx.x, wid = tid >> 5;
  int wm = wid >> 2, wn = wid & 3;
  v8f acc0 = {}; v8f acc1 = {};

  // per-thread staging coordinates (8 f16 = 16 B per thread per tile)
  const int ar = tid >> 2, ac = (tid & 3) * 8;   // A: 64 rows x 4 segs
  const int br = tid >> 3, bc = (tid & 7) * 8;   // B: 32 rows x 8 segs

  for (int k0 = 0; k0 < K; k0 += 32) {
    // async global -> LDS staging (no VGPR round-trip)
    async_copy_b128(&As[ar * 32 + ac], A + (size_t)(bm * 64 + ar) * K + k0 + ac);
    async_copy_b128(&Bs[br * 64 + bc], B + (size_t)(k0 + br) * N + bn * 64 + bc);
    if (k0 + 32 < K) {  // pull next K-tile toward L2 (global_prefetch_b8)
      __builtin_prefetch(A + (size_t)(bm * 64 + ar) * K + k0 + 32, 0, 1);
      __builtin_prefetch(B + (size_t)(k0 + 32 + br) * N + bn * 64, 0, 1);
    }
    async_wait_all();
    __syncthreads();
    v16h af0 = load_frag_a(As, 32, wm * 32, 0);
    v16h af1 = load_frag_a(As, 32, wm * 32 + 16, 0);
    v16h bf  = load_frag_b(Bs, 64, 0, wn * 16);
    acc0 = WMMA_F16(af0, bf, acc0);
    acc1 = WMMA_F16(af1, bf, acc1);
    __syncthreads();
  }

  int l = lane_id();
  int nc = bn * 64 + wn * 16 + (l & 15);
  int half = l >> 4;
  float bv = bias ? bias[nc] : 0.f;
#pragma unroll
  for (int i = 0; i < 8; ++i) {
    int r0 = bm * 64 + wm * 32 + i + 8 * half;
    float v0 = acc0[i] + bv;
    float v1 = acc1[i] + bv;
    if (act == 1) { v0 = gelu_tanh(v0); v1 = gelu_tanh(v1); }
    C[(size_t)r0 * N + nc] = v0;
    C[(size_t)(r0 + 16) * N + nc] = v1;
  }
}

// ---------------------------------------------------------------------------
// Small GEMM (N in {12,60}): one thread per output. act: 0 none, 2 sigmoid.
// ---------------------------------------------------------------------------
__global__ void small_gemm_kernel(const float* __restrict__ A, const float* __restrict__ B,
                                  const float* __restrict__ bias, float* __restrict__ O,
                                  int M, int K, int N, int act) {
  int idx = blockIdx.x * blockDim.x + threadIdx.x;
  if (idx >= M * N) return;
  int m = idx / N, n = idx - m * N;
  float s = bias ? bias[n] : 0.f;
  for (int k = 0; k < K; ++k) s += A[(size_t)m * K + k] * B[(size_t)k * N + n];
  if (act == 2) s = 1.f / (1.f + __expf(-s));
  O[idx] = s;
}

// ---------------------------------------------------------------------------
// Chunkwise delta rule, one block per (b,h). beta == 1 baked in.
// Per chunk (c=64, D=64):
//   k <- k / (||k|| + 1e-6); q <- q * D^-1/2
//   A  = tril(k k^T, -1)                       [WMMA]
//   T  = (I + A)^-1                            [serial forward substitution]
//   X  = v - k @ S                             [WMMA, C loaded from global v]
//   ut = T @ X                                 [WMMA]
//   at = tril(q k^T)                           [WMMA]
//   o  = q @ S + at @ ut                       [WMMA, chained accumulators]
//   S += k^T @ ut                              [WMMA, C loaded from LDS S]
// ---------------------------------------------------------------------------
__global__ __launch_bounds__(256) void delta_kernel(const float* __restrict__ qkv,
                                                    float* __restrict__ dout) {
  __shared__ float    S_[64 * 64];
  __shared__ float    Am[64 * 64];
  __shared__ float    Tm[64 * 64];
  __shared__ _Float16 qh[64 * 64];
  __shared__ _Float16 kh[64 * 64];
  __shared__ _Float16 nkh[64 * 64];
  __shared__ _Float16 Th[64 * 64];
  __shared__ _Float16 Sh[64 * 64];
  __shared__ _Float16 Xh[64 * 64];
  __shared__ _Float16 uth[64 * 64];
  __shared__ _Float16 ah[64 * 64];

  const int b = blockIdx.x / 12, h = blockIdx.x % 12;
  const int tid = threadIdx.x, wid = tid >> 5, l = tid & 31;
  const float scale = 0.125f;  // 64^-0.5

  for (int i = tid; i < 4096; i += 256) S_[i] = 0.f;
  __syncthreads();

  for (int ci = 0; ci < 16; ++ci) {
    const float* base = qkv + ((size_t)(b * 1024 + ci * 64)) * 2304 + h * 64;

    // ---- load + normalize (64 threads, one row each) ----
    if (tid < 64) {
      const float* qr = base + (size_t)tid * 2304;
      const float* kr = qr + 768;
      float ss = 0.f;
      for (int d = 0; d < 64; ++d) { float kd = kr[d]; ss += kd * kd; }
      float inv = 1.f / (sqrtf(ss) + 1e-6f);
      for (int d = 0; d < 64; ++d) {
        float kd = kr[d] * inv;
        kh[tid * 64 + d]  = (_Float16)kd;
        nkh[tid * 64 + d] = (_Float16)(-kd);
        qh[tid * 64 + d]  = (_Float16)(qr[d] * scale);
      }
    }
    __syncthreads();

    // ---- A = tril(k k^T, -1) ----
    for (int tt = wid * 2; tt < wid * 2 + 2; ++tt) {
      int tm = (tt >> 2) * 16, tn = (tt & 3) * 16;
      v8f acc = {};
      acc = WMMA_F16(load_frag_a(kh, 64, tm, 0),  load_frag_bt(kh, 64, 0, tn),  acc);
      acc = WMMA_F16(load_frag_a(kh, 64, tm, 32), load_frag_bt(kh, 64, 32, tn), acc);
      int nc = tn + (l & 15), half = l >> 4;
#pragma unroll
      for (int i = 0; i < 8; ++i) {
        int r = tm + i + 8 * half;
        Am[r * 64 + nc] = (r > nc) ? acc[i] : 0.f;
      }
    }
    __syncthreads();

    // ---- T = (I+A)^-1 by forward substitution ----
    for (int i = tid; i < 4096; i += 256) Tm[i] = ((i >> 6) == (i & 63)) ? 1.f : 0.f;
    __syncthreads();
    for (int i = 1; i < 64; ++i) {
      if (tid < i) {
        float s2 = 0.f;
        for (int j = tid; j < i; ++j) s2 += Am[i * 64 + j] * Tm[j * 64 + tid];
        Tm[i * 64 + tid] = -s2;
      }
      __syncthreads();
    }
    for (int i = tid; i < 4096; i += 256) {
      Th[i] = (_Float16)Tm[i];
      Sh[i] = (_Float16)S_[i];
    }
    __syncthreads();

    // ---- X = v - k @ S  (C fragment loaded straight from global v) ----
    const float* vbase = base + 1536;
    for (int tt = wid * 2; tt < wid * 2 + 2; ++tt) {
      int tm = (tt >> 2) * 16, tn = (tt & 3) * 16;
      int nc = tn + (l & 15), half = l >> 4;
      v8f acc;
#pragma unroll
      for (int i = 0; i < 8; ++i) acc[i] = vbase[(size_t)(tm + i + 8 * half) * 2304 + nc];
      acc = WMMA_F16(load_frag_a(nkh, 64, tm, 0),  load_frag_b(Sh, 64, 0, tn),  acc);
      acc = WMMA_F16(load_frag_a(nkh, 64, tm, 32), load_frag_b(Sh, 64, 32, tn), acc);
#pragma unroll
      for (int i = 0; i < 8; ++i) Xh[(tm + i + 8 * half) * 64 + nc] = (_Float16)acc[i];
    }
    __syncthreads();

    // ---- ut = T @ X ----
    for (int tt = wid * 2; tt < wid * 2 + 2; ++tt) {
      int tm = (tt >> 2) * 16, tn = (tt & 3) * 16;
      v8f acc = {};
      acc = WMMA_F16(load_frag_a(Th, 64, tm, 0),  load_frag_b(Xh, 64, 0, tn),  acc);
      acc = WMMA_F16(load_frag_a(Th, 64, tm, 32), load_frag_b(Xh, 64, 32, tn), acc);
      int nc = tn + (l & 15), half = l >> 4;
#pragma unroll
      for (int i = 0; i < 8; ++i) uth[(tm + i + 8 * half) * 64 + nc] = (_Float16)acc[i];
    }
    __syncthreads();

    // ---- at = tril(q k^T) ----
    for (int tt = wid * 2; tt < wid * 2 + 2; ++tt) {
      int tm = (tt >> 2) * 16, tn = (tt & 3) * 16;
      v8f acc = {};
      acc = WMMA_F16(load_frag_a(qh, 64, tm, 0),  load_frag_bt(kh, 64, 0, tn),  acc);
      acc = WMMA_F16(load_frag_a(qh, 64, tm, 32), load_frag_bt(kh, 64, 32, tn), acc);
      int nc = tn + (l & 15), half = l >> 4;
#pragma unroll
      for (int i = 0; i < 8; ++i) {
        int r = tm + i + 8 * half;
        ah[r * 64 + nc] = (r >= nc) ? (_Float16)acc[i] : (_Float16)0.f;
      }
    }
    __syncthreads();

    // ---- o = q @ S + at @ ut  -> global ----
    for (int tt = wid * 2; tt < wid * 2 + 2; ++tt) {
      int tm = (tt >> 2) * 16, tn = (tt & 3) * 16;
      v8f acc = {};
      acc = WMMA_F16(load_frag_a(qh, 64, tm, 0),  load_frag_b(Sh, 64, 0, tn),  acc);
      acc = WMMA_F16(load_frag_a(qh, 64, tm, 32), load_frag_b(Sh, 64, 32, tn), acc);
      acc = WMMA_F16(load_frag_a(ah, 64, tm, 0),  load_frag_b(uth, 64, 0, tn),  acc);
      acc = WMMA_F16(load_frag_a(ah, 64, tm, 32), load_frag_b(uth, 64, 32, tn), acc);
      int nc = tn + (l & 15), half = l >> 4;
#pragma unroll
      for (int i = 0; i < 8; ++i) {
        int r = tm + i + 8 * half;
        dout[(size_t)(b * 1024 + ci * 64 + r) * 768 + h * 64 + nc] = acc[i];
      }
    }
    __syncthreads();

    // ---- S += k^T @ ut ----
    for (int tt = wid * 2; tt < wid * 2 + 2; ++tt) {
      int tm = (tt >> 2) * 16, tn = (tt & 3) * 16;
      int nc = tn + (l & 15), half = l >> 4;
      v8f acc;
#pragma unroll
      for (int i = 0; i < 8; ++i) acc[i] = S_[(tm + i + 8 * half) * 64 + nc];
      acc = WMMA_F16(load_frag_at(kh, 64, tm, 0),  load_frag_b(uth, 64, 0, tn),  acc);
      acc = WMMA_F16(load_frag_at(kh, 64, tm, 32), load_frag_b(uth, 64, 32, tn), acc);
#pragma unroll
      for (int i = 0; i < 8; ++i) S_[(tm + i + 8 * half) * 64 + nc] = acc[i];
    }
    __syncthreads();
  }
}

// ---------------------------------------------------------------------------
// Fused FIR + head-mix + path-prune-softmax + gating + entropy, one (b,n)/block.
// Produces context in f16 for the final WMMA GEMM, accumulates entropy sums.
// ---------------------------------------------------------------------------
__global__ __launch_bounds__(256) void fuse_kernel(
    const float* __restrict__ qkv, const float* __restrict__ deltao,
    const float* __restrict__ idraw, const float* __restrict__ prune,
    const float* __restrict__ gate, const float* __restrict__ idgate,
    const float* __restrict__ firs, const float* __restrict__ firl,
    const float* __restrict__ mixm, const float* __restrict__ id_static,
    _Float16* __restrict__ ctxh, float* __restrict__ eacc) {
  const int row = blockIdx.x;  // b*1024 + n
  const int b = row >> 10, n = row & 1023;
  __shared__ float vbuf[7][768];  // v at n-6 .. n
  __shared__ float fsb[768], flb[768], mxs[768], mxl[768];
  __shared__ float wls[12][5];
  __shared__ float gsh[12], dgsh[12];
  __shared__ float e1s, e2s;
  const int tid = threadIdx.x;

  for (int t = 0; t < 7; ++t) {
    int nn = n - 6 + t;
    for (int i = tid; i < 768; i += 256)
      vbuf[t][i] = (nn >= 0) ? qkv[((size_t)(b * 1024 + nn)) * 2304 + 1536 + i] : 0.f;
  }
  __syncthreads();

  // causal FIR: y[n] = sum_t w[t] * x[n-(K-1)+t]
  for (int i = tid; i < 768; i += 256) {
    float a = 0.f;
#pragma unroll
    for (int t = 0; t < 3; ++t) a += firs[i * 3 + t] * vbuf[4 + t][i];
    float c7 = 0.f;
#pragma unroll
    for (int t = 0; t < 7; ++t) c7 += firl[i * 7 + t] * vbuf[t][i];
    fsb[i] = a; flb[i] = c7;
  }
  __syncthreads();

  // residual head mixing: mx[g,d] = fir[g,d] + sum_h fir[h,d]*mix[h,g]
  for (int i = tid; i < 768; i += 256) {
    int g = i >> 6, d = i & 63;
    float s1 = fsb[i], s2 = flb[i];
    for (int hh = 0; hh < 12; ++hh) {
      float mv = mixm[hh * 12 + g];
      s1 += fsb[hh * 64 + d] * mv;
      s2 += flb[hh * 64 + d] * mv;
    }
    mxs[i] = s1; mxl[i] = s2;
  }

  if (tid == 0) { e1s = 0.f; e2s = 0.f; }
  if (tid < 12) {  // per-head softmax over P=5 paths + gates + entropies (wave 0)
    float lg[5];
    float mx = -1e30f;
    for (int p = 0; p < 5; ++p) { lg[p] = prune[(size_t)row * 60 + tid * 5 + p]; mx = fmaxf(mx, lg[p]); }
    float sum = 0.f;
    for (int p = 0; p < 5; ++p) { lg[p] = __expf(lg[p] - mx); sum += lg[p]; }
    float inv = 1.f / sum;
    float e1 = 0.f;
    for (int p = 0; p < 5; ++p) {
      float w = lg[p] * inv;
      wls[tid][p] = w;
      e1 += w * __logf(w + 1e-8f);
    }
    float g = gate[(size_t)row * 12 + tid];
    gsh[tid] = g;
    float st = 1.f / (1.f + __expf(-id_static[tid]));
    dgsh[tid] = idgate[(size_t)row * 12 + tid] * st;
    float e2 = g * __logf(g + 1e-8f) + (1.f - g) * __logf(1.f - g + 1e-8f);
    atomicAdd(&e1s, e1);
    atomicAdd(&e2s, e2);
  }
  __syncthreads();

  for (int i = tid; i < 768; i += 256) {
    int hh = i >> 6;
    float p2 = deltao[(size_t)row * 768 + i];
    float p3 = vbuf[6][i];
    float p4 = idraw[(size_t)row * 768 + i] * dgsh[hh];
    float pr = wls[hh][0] * mxs[i] + wls[hh][1] * mxl[i] + wls[hh][2] * p2 +
               wls[hh][3] * p3 + wls[hh][4] * p4;
    ctxh[(size_t)row * 768 + i] = (_Float16)(pr * gsh[hh]);
  }
  if (tid == 0) { atomicAdd(&eacc[0], e1s); atomicAdd(&eacc[1], e2s); }
}

__global__ void init_eacc_kernel(float* eacc) {
  if (threadIdx.x < 2) eacc[threadIdx.x] = 0.f;
}

__global__ void finalize_kernel(const float* eacc, float* out_scalar) {
  if (threadIdx.x == 0) out_scalar[0] = -(eacc[0] + eacc[1]) / 49152.f;  // mean over B*N*H
}

// ---------------------------------------------------------------------------
extern "C" void kernel_launch(void* const* d_in, const int* in_sizes, int n_in,
                              void* d_out, int out_size, void* d_ws, size_t ws_size,
                              hipStream_t stream) {
  const float* x         = (const float*)d_in[0];   // [4,1024,768]
  const float* w_qkv     = (const float*)d_in[1];   // [768,2304]
  const float* w_proj    = (const float*)d_in[2];   // [768,768]
  const float* b_proj    = (const float*)d_in[3];   // [768]
  const float* fir_s     = (const float*)d_in[4];   // [12,64,3]
  const float* fir_l     = (const float*)d_in[5];   // [12,64,7]
  const float* mixm      = (const float*)d_in[6];   // [12,12]
  const float* w_prune   = (const float*)d_in[7];   // [768,60]
  const float* b_prune   = (const float*)d_in[8];   // [60]
  const float* w_g1      = (const float*)d_in[9];   // [768,768]
  const float* b_g1      = (const float*)d_in[10];  // [768]
  const float* w_g2      = (const float*)d_in[11];  // [768,12]
  const float* b_g2      = (const float*)d_in[12];  // [12]
  const float* w_id      = (const float*)d_in[13];  // [768,768]
  const float* id_static = (const float*)d_in[14];  // [12]
  const float* w_idgate  = (const float*)d_in[15];  // [768,12]
  const float* b_idgate  = (const float*)d_in[16];  // [12]
  float* out = (float*)d_out;  // 4*1024*768 floats + 1 scalar

  const int BN = 4096, C = 768, CQ = 2304;

  // workspace carve-out (~96 MB)
  size_t off = 0;
  auto take = [&](size_t bytes) -> void* {
    void* p = (char*)d_ws + off;
    off += (bytes + 255) & ~(size_t)255;
    return p;
  };
  _Float16* xh     = (_Float16*)take((size_t)BN * C * 2);
  _Float16* wqkvh  = (_Float16*)take((size_t)C * CQ * 2);
  _Float16* widh   = (_Float16*)take((size_t)C * C * 2);
  _Float16* wg1h   = (_Float16*)take((size_t)C * C * 2);
  _Float16* wprojh = (_Float16*)take((size_t)C * C * 2);
  _Float16* ctxh   = (_Float16*)take((size_t)BN * C * 2);
  float* qkvb   = (float*)take((size_t)BN * CQ * 4);
  float* idraw  = (float*)take((size_t)BN * C * 4);
  float* hdn    = (float*)take((size_t)BN * C * 4);
  float* deltao = (float*)take((size_t)BN * C * 4);
  float* prune  = (float*)take((size_t)BN * 60 * 4);
  float* gateb  = (float*)take((size_t)BN * 12 * 4);
  float* idg    = (float*)take((size_t)BN * 12 * 4);
  float* eacc   = (float*)take(2 * 4);
  (void)ws_size; (void)in_sizes; (void)n_in; (void)out_size;

  auto cvt = [&](const float* src, _Float16* dst, int n) {
    cvt_f16_kernel<<<(n + 255) / 256, 256, 0, stream>>>(src, dst, n);
  };
  cvt(x, xh, BN * C);
  cvt(w_qkv, wqkvh, C * CQ);
  cvt(w_id, widh, C * C);
  cvt(w_g1, wg1h, C * C);
  cvt(w_proj, wprojh, C * C);
  init_eacc_kernel<<<1, 32, 0, stream>>>(eacc);

  // big WMMA GEMMs
  gemm_f16_wmma<<<dim3(CQ / 64, BN / 64), 256, 0, stream>>>(xh, wqkvh, qkvb, nullptr, BN, C, CQ, 0);
  gemm_f16_wmma<<<dim3(C / 64, BN / 64), 256, 0, stream>>>(xh, widh, idraw, nullptr, BN, C, C, 0);
  gemm_f16_wmma<<<dim3(C / 64, BN / 64), 256, 0, stream>>>(xh, wg1h, hdn, b_g1, BN, C, C, 1);

  // small GEMMs (N=60/12)
  small_gemm_kernel<<<(BN * 60 + 255) / 256, 256, 0, stream>>>(x, w_prune, b_prune, prune, BN, C, 60, 0);
  small_gemm_kernel<<<(BN * 12 + 255) / 256, 256, 0, stream>>>(hdn, w_g2, b_g2, gateb, BN, C, 12, 2);
  small_gemm_kernel<<<(BN * 12 + 255) / 256, 256, 0, stream>>>(x, w_idgate, b_idgate, idg, BN, C, 12, 2);

  // chunkwise delta rule, one block per (b,h)
  delta_kernel<<<48, 256, 0, stream>>>(qkvb, deltao);

  // FIR + mix + prune + gate + entropies -> context (f16)
  fuse_kernel<<<BN, 256, 0, stream>>>(qkvb, deltao, idraw, prune, gateb, idg,
                                      fir_s, fir_l, mixm, id_static, ctxh, eacc);

  // final projection straight into d_out
  gemm_f16_wmma<<<dim3(C / 64, BN / 64), 256, 0, stream>>>(ctxh, wprojh, out, b_proj, BN, C, C, 0);
  finalize_kernel<<<1, 32, 0, stream>>>(eacc, out + (size_t)BN * C);
}
